// SmolLM2Attention135M_60868276519096
// MI455X (gfx1250) — matrix operand order repaired and checked by reading, not compiled, not run on hardware
//
#include <hip/hip_runtime.h>
#include <math.h>

// ---------------------------------------------------------------------------
// SmolLM2-135M attention for MI455X (gfx1250, wave32).
// All matmuls: V_WMMA_F32_16X16X32_BF16 (bf16 A/B, fp32 accumulate).
// Pipeline: [convert x/W to bf16] -> [qkv+rope] -> [flash causal attention,
//           64 kv per step, staged fragments] -> [output projection (fp32)]
// ---------------------------------------------------------------------------

typedef __bf16 bf16_t;
typedef __attribute__((ext_vector_type(16))) __bf16 v16bf;
typedef __attribute__((ext_vector_type(8)))  __bf16 v8bf;
typedef __attribute__((ext_vector_type(8)))  float  v8f;

#define HIDDEN   576
#define NH       9
#define NKV      3
#define HD       64
#define BATCH    4
#define SEQ      2048
#define REP      3
#define QKV_COLS 960   // 576 q | 192 k | 192 v fused
#define PROW     72    // padded LDS row stride (bf16) -> conflict-free b128

__device__ __forceinline__ v16bf join8(v8bf lo, v8bf hi) {
  union { v16bf v; v8bf h[2]; } u;
  u.h[0] = lo; u.h[1] = hi;
  return u.v;
}

__device__ __forceinline__ v8f wmma_bf16(v16bf a, v16bf b, v8f c) {
  // 8 args: (neg_a, A, neg_b, B, c_mod, C, reuse_a, reuse_b)
  return __builtin_amdgcn_wmma_f32_16x16x32_bf16(false, a, false, b, (short)0,
                                                 c, false, false);
}

// A-matrix 16x32 bf16 fragment (ISA 7.12.2): lane row M = lane%16,
// e0..7 = K (hi*8 .. +7), e8..15 = K (16+hi*8 .. +7). Two contiguous 16B loads.
__device__ __forceinline__ v16bf load_afrag(const bf16_t* base, int hi) {
  v8bf lo = *(const v8bf*)(base + hi * 8);
  v8bf hh = *(const v8bf*)(base + 16 + hi * 8);
  return join8(lo, hh);
}
// B-matrix 32x16 bf16 fragment: lane col fixed, e0..7 = K (hi*8 .. +7), e8..15 = K (16+hi*8 .. +7).
__device__ __forceinline__ v16bf load_bfrag(const bf16_t* base, int hi) {
  v8bf lo = *(const v8bf*)(base + hi * 8);
  v8bf hh = *(const v8bf*)(base + 16 + hi * 8);
  return join8(lo, hh);
}
// Same A fragment shape from LDS.
__device__ __forceinline__ v16bf lds_afrag(const bf16_t* base, int hi) {
  v8bf lo = *(const v8bf*)(base + hi * 8);
  v8bf hh = *(const v8bf*)(base + 16 + hi * 8);
  return join8(lo, hh);
}

// ---------------------------------------------------------------------------
// Conversion kernels (one-time per call; ~30MB traffic total).
// ---------------------------------------------------------------------------
__global__ __launch_bounds__(256)
void cvt_f32_bf16(const float* __restrict__ in, bf16_t* __restrict__ out, int n) {
  int i = blockIdx.x * blockDim.x + threadIdx.x;
  if (i < n) out[i] = (bf16_t)in[i];
}

__global__ __launch_bounds__(256)
void cvt_wqkv(const float* __restrict__ Wq, const float* __restrict__ Wk,
              const float* __restrict__ Wv, bf16_t* __restrict__ WT) {
  int idx = blockIdx.x * blockDim.x + threadIdx.x;   // WT[col][k], col<960
  if (idx >= QKV_COLS * HIDDEN) return;
  int col = idx / HIDDEN, k = idx - col * HIDDEN;
  float v;
  if (col < 576)      v = Wq[(size_t)k * 576 + col];
  else if (col < 768) v = Wk[(size_t)k * 192 + (col - 576)];
  else                v = Wv[(size_t)k * 192 + (col - 768)];
  WT[idx] = (bf16_t)v;
}

__global__ __launch_bounds__(256)
void cvt_wo(const float* __restrict__ Wo, bf16_t* __restrict__ WoT) {
  int idx = blockIdx.x * blockDim.x + threadIdx.x;   // WoT[col][k]
  if (idx >= HIDDEN * HIDDEN) return;
  int col = idx / HIDDEN, k = idx - col * HIDDEN;
  WoT[idx] = (bf16_t)Wo[(size_t)k * HIDDEN + col];
}

// ---------------------------------------------------------------------------
// Kernel 1: fused QKV projection + RoPE.  MODE: 0=Q (paired tiles, rope),
// 1=K (paired, rope), 2=V (single tile). Mode is a template param so the
// inner loop is branch-free.
//   Q -> qb [B,NH,S,HD] bf16      (natural: A-frags contiguous)
//   K -> kb [B,NKV,S,HD] bf16     (natural: QK^T B-frags contiguous in d)
//   V -> vT [B,NKV,HD,S] bf16     (transposed: P@V B-frags contiguous in kv)
// ---------------------------------------------------------------------------
template <int MODE>
__global__ __launch_bounds__(32)
void qkv_kernel(const bf16_t* __restrict__ xb,
                const bf16_t* __restrict__ WT,
                bf16_t* __restrict__ qb,
                bf16_t* __restrict__ kb,
                bf16_t* __restrict__ vT) {
  const int lane = threadIdx.x;
  const int lm = lane & 15, hi = lane >> 4;
  const int rowBase = blockIdx.x * 16;
  const int job = blockIdx.y;

  int head, csel, wcol0;
  if (MODE == 0)      { head = job >> 1; csel = (job & 1) * 16; wcol0 = head * HD + csel; }
  else if (MODE == 1) { head = job >> 1; csel = (job & 1) * 16; wcol0 = 576 + head * HD + csel; }
  else                { head = (job * 16) / HD; csel = 0; wcol0 = 768 + job * 16; }

  const bf16_t* arow  = xb + (size_t)(rowBase + lm) * HIDDEN;
  const bf16_t* b0row = WT + (size_t)(wcol0 + lm) * HIDDEN;
  const bf16_t* b1row = b0row + (size_t)32 * HIDDEN;

  v8f acc0 = {}, acc1 = {};
#pragma unroll 3
  for (int kk = 0; kk < HIDDEN; kk += 32) {
    v16bf a  = load_afrag(arow + kk, hi);
    v16bf b0 = load_bfrag(b0row + kk, hi);
    acc0 = wmma_bf16(a, b0, acc0);
    if (MODE != 2) {
      v16bf b1 = load_bfrag(b1row + kk, hi);
      acc1 = wmma_bf16(a, b1, acc1);
    }
  }

  if (MODE == 2) {
    const int d = (job * 16) % HD + lm;
    for (int r = 0; r < 8; ++r) {
      int tok = rowBase + r + hi * 8;
      int b = tok >> 11, pos = tok & (SEQ - 1);
      vT[(((size_t)b * NKV + head) * HD + d) * SEQ + pos] = (bf16_t)acc0[r];
    }
  } else {
    // RoPE: pair (d0, d0+32) lives in (acc0[r], acc1[r]) of the same lane.
    const int d0 = csel + lm;                               // [0,32)
    const float freq = __expf(-0.28782313662425572f * (float)d0); // 10000^(-d0/32)
    for (int r = 0; r < 8; ++r) {
      int tok = rowBase + r + hi * 8;
      int b = tok >> 11, pos = tok & (SEQ - 1);
      float sv, cv;
      sincosf((float)pos * freq, &sv, &cv);
      float o0 = acc0[r] * cv - acc1[r] * sv;
      float o1 = acc1[r] * cv + acc0[r] * sv;
      if (MODE == 0) {
        size_t base = (((size_t)b * NH + head) * SEQ + pos) * HD;
        qb[base + d0]      = (bf16_t)o0;
        qb[base + d0 + 32] = (bf16_t)o1;
      } else {
        size_t base = (((size_t)b * NKV + head) * SEQ + pos) * HD;
        kb[base + d0]      = (bf16_t)o0;
        kb[base + d0 + 32] = (bf16_t)o1;
      }
    }
  }
}

// ---------------------------------------------------------------------------
// Kernel 2: flash causal attention. One wave per (16-query tile, head, batch).
// 64 kv per step. All K AND V fragments for the slab are loaded into distinct
// registers up front (one clause, one wait for K; V latency hides behind the
// softmax), then 8 QK^T wmma + softmax + LDS transpose + 8 P@V wmma.
// ---------------------------------------------------------------------------
__global__ __launch_bounds__(32)
void attn_kernel(const bf16_t* __restrict__ qb,   // [B,NH,S,HD]
                 const bf16_t* __restrict__ kb,   // [B,NKV,S,HD]
                 const bf16_t* __restrict__ vT,   // [B,NKV,HD,S]
                 bf16_t* __restrict__ ctx) {      // [B*S, HIDDEN] bf16
  __shared__ bf16_t pl[16 * PROW];
  const int lane = threadIdx.x;
  const int lm = lane & 15, hi = lane >> 4;
  const int qt = blockIdx.x, h = blockIdx.y, b = blockIdx.z;
  const int hkv = h / REP;
  const int qBase = qt * 16;

  const bf16_t* qrow = qb + (((size_t)b * NH + h) * SEQ + qBase + lm) * HD;
  v16bf aq0 = load_afrag(qrow, hi);        // d = 0..31
  v16bf aq1 = load_afrag(qrow + 32, hi);   // d = 32..63

  const bf16_t* kbase = kb + ((size_t)b * NKV + hkv) * (size_t)SEQ * HD;
  const bf16_t* vbase = vT + ((size_t)b * NKV + hkv) * (size_t)HD * SEQ;

  float m[8], l[8], corr[8];
  v8f acc[4] = {};
  for (int r = 0; r < 8; ++r) { m[r] = -1e30f; l[r] = 0.0f; }

  const int kvLimit = qBase + 16;
  const int nblk  = (kvLimit + 63) / 64;                          // 64-kv slabs
  const int nfull = (qBase >= 63) ? ((qBase - 63) / 64 + 1) : 0;  // no-mask slabs

  auto process = [&](int kvBase, bool domask) {
    // ---- stage ALL fragments for this slab into registers ----
    v16bf kf[8], vf[8];
#pragma unroll
    for (int t = 0; t < 4; ++t) {
      const bf16_t* kt = kbase + (size_t)(kvBase + t * 16 + lm) * HD;
      kf[2 * t]     = load_bfrag(kt, hi);
      kf[2 * t + 1] = load_bfrag(kt + 32, hi);
    }
#pragma unroll
    for (int t = 0; t < 4; ++t) {
      const bf16_t* vp = vbase + (size_t)(t * 16 + lm) * SEQ + kvBase;
      vf[2 * t]     = load_bfrag(vp, hi);      // consumed only after softmax:
      vf[2 * t + 1] = load_bfrag(vp + 32, hi); // latency hidden by VALU work
    }
    // Prefetch next slab's K/V rows (L2 -> WGP) while this slab computes.
    if (kvBase + 64 < kvLimit) {
#pragma unroll
      for (int t = 0; t < 4; ++t) {
        __builtin_prefetch(kbase + (size_t)(kvBase + 64 + t * 16 + lm) * HD, 0, 3);
        __builtin_prefetch(vbase + (size_t)(t * 16 + lm) * SEQ + kvBase + 64, 0, 3);
      }
    }

    // ---- scores for kv columns [kvBase, kvBase+64): 4 tiles x 2 wmma ----
    v8f sc[4];
#pragma unroll
    for (int t = 0; t < 4; ++t) {
      v8f s = {};
      s = wmma_bf16(aq0, kf[2 * t], s);
      s = wmma_bf16(aq1, kf[2 * t + 1], s);
      sc[t] = s;
    }

    float p[4][8];
#pragma unroll
    for (int t = 0; t < 4; ++t)
      for (int r = 0; r < 8; ++r) {
        float s = sc[t][r] * 0.125f;       // 1/sqrt(64)
        if (domask) {
          int qidx = qBase + r + hi * 8;
          if (kvBase + t * 16 + lm > qidx) s = -1e30f;
        }
        p[t][r] = s;
      }

    // ---- online softmax: per-row reductions across the 16-lane half ----
#pragma unroll
    for (int r = 0; r < 8; ++r) {
      float tm = fmaxf(fmaxf(p[0][r], p[1][r]), fmaxf(p[2][r], p[3][r]));
      for (int off = 8; off >= 1; off >>= 1)
        tm = fmaxf(tm, __shfl_xor(tm, off, 32));
      float mn = fmaxf(m[r], tm);
      corr[r] = __expf(m[r] - mn);
      m[r] = mn;
      float rs = 0.0f;
#pragma unroll
      for (int t = 0; t < 4; ++t) {
        p[t][r] = __expf(p[t][r] - mn);
        rs += p[t][r];
      }
      for (int off = 8; off >= 1; off >>= 1)
        rs += __shfl_xor(rs, off, 32);
      l[r] = l[r] * corr[r] + rs;
    }
#pragma unroll
    for (int t = 0; t < 4; ++t)
      for (int r = 0; r < 8; ++r) acc[t][r] *= corr[r];

    // ---- P (C-layout fp32) -> bf16 LDS [16 rows][64+pad] -> A-frags ----
#pragma unroll
    for (int r = 0; r < 8; ++r) {
      int row = r + hi * 8;
#pragma unroll
      for (int t = 0; t < 4; ++t)
        pl[row * PROW + t * 16 + lm] = (bf16_t)p[t][r];
    }
    __syncthreads();
    v16bf ap0 = lds_afrag(&pl[lm * PROW], hi);        // kv 0..31 of slab
    v16bf ap1 = lds_afrag(&pl[lm * PROW + 32], hi);   // kv 32..63 of slab
    __syncthreads();

    // ---- acc += P @ V : 4 d-tiles x 2 kv chunks, operands already resident ----
#pragma unroll
    for (int t = 0; t < 4; ++t) {
      acc[t] = wmma_bf16(ap0, vf[2 * t], acc[t]);
      acc[t] = wmma_bf16(ap1, vf[2 * t + 1], acc[t]);
    }
  };

  for (int j = 0; j < nfull; ++j) process(j * 64, false);
  for (int j = nfull; j < nblk; ++j) process(j * 64, true);

  for (int r = 0; r < 8; ++r) {
    int M = r + hi * 8;
    size_t tok = (size_t)b * SEQ + qBase + M;
    float inv = 1.0f / l[r];
    for (int t = 0; t < 4; ++t)
      ctx[tok * HIDDEN + h * HD + t * 16 + lm] = (bf16_t)(acc[t][r] * inv);
  }
}

// ---------------------------------------------------------------------------
// Kernel 3: output projection ctx(bf16)[B*S,576] @ Wo -> out fp32.
// ---------------------------------------------------------------------------
__global__ __launch_bounds__(32)
void oproj_kernel(const bf16_t* __restrict__ ctx,
                  const bf16_t* __restrict__ WoT,   // [576 cols][576 k]
                  float* __restrict__ out) {
  const int lane = threadIdx.x;
  const int lm = lane & 15, hi = lane >> 4;
  const int rowBase = blockIdx.x * 16;
  const int n0 = blockIdx.y * 16;
  const bf16_t* arow = ctx + (size_t)(rowBase + lm) * HIDDEN;
  const bf16_t* brow = WoT + (size_t)(n0 + lm) * HIDDEN;
  v8f acc = {};
#pragma unroll 3
  for (int kk = 0; kk < HIDDEN; kk += 32)
    acc = wmma_bf16(load_afrag(arow + kk, hi), load_bfrag(brow + kk, hi), acc);
  for (int r = 0; r < 8; ++r) {
    int row = rowBase + r + hi * 8;
    out[(size_t)row * HIDDEN + n0 + lm] = acc[r];
  }
}

// ---------------------------------------------------------------------------
extern "C" void kernel_launch(void* const* d_in, const int* in_sizes, int n_in,
                              void* d_out, int out_size, void* d_ws,
                              size_t ws_size, hipStream_t stream) {
  (void)in_sizes; (void)n_in; (void)out_size; (void)ws_size;
  const float* x  = (const float*)d_in[0];
  const float* Wq = (const float*)d_in[1];
  const float* Wk = (const float*)d_in[2];
  const float* Wv = (const float*)d_in[3];
  const float* Wo = (const float*)d_in[4];
  float* out = (float*)d_out;

  // Workspace (bf16): xb | WT | WoT | qb | kb | vT | ctx  (~36.4 MB)
  bf16_t* p = (bf16_t*)d_ws;
  bf16_t* xb  = p; p += (size_t)BATCH * SEQ * HIDDEN;
  bf16_t* WT  = p; p += (size_t)QKV_COLS * HIDDEN;
  bf16_t* WoT = p; p += (size_t)HIDDEN * HIDDEN;
  bf16_t* qb  = p; p += (size_t)BATCH * NH  * SEQ * HD;
  bf16_t* kb  = p; p += (size_t)BATCH * NKV * SEQ * HD;
  bf16_t* vT  = p; p += (size_t)BATCH * NKV * HD  * SEQ;
  bf16_t* ctx = p;

  const int n_x = BATCH * SEQ * HIDDEN;
  cvt_f32_bf16<<<(n_x + 255) / 256, 256, 0, stream>>>(x, xb, n_x);
  cvt_wqkv<<<(QKV_COLS * HIDDEN + 255) / 256, 256, 0, stream>>>(Wq, Wk, Wv, WT);
  cvt_wo<<<(HIDDEN * HIDDEN + 255) / 256, 256, 0, stream>>>(Wo, WoT);

  dim3 gq((BATCH * SEQ) / 16, 18);
  dim3 gk((BATCH * SEQ) / 16, 6);
  dim3 gv((BATCH * SEQ) / 16, 12);
  qkv_kernel<0><<<gq, 32, 0, stream>>>(xb, WT, qb, kb, vT);
  qkv_kernel<1><<<gk, 32, 0, stream>>>(xb, WT, qb, kb, vT);
  qkv_kernel<2><<<gv, 32, 0, stream>>>(xb, WT, qb, kb, vT);

  dim3 ga(SEQ / 16, NH, BATCH);
  attn_kernel<<<ga, 32, 0, stream>>>(qb, kb, vT, ctx);

  dim3 go((BATCH * SEQ) / 16, HIDDEN / 16);
  oproj_kernel<<<go, 32, 0, stream>>>(ctx, WoT, out);
}
